// AttentionCell_2576980377639
// MI455X (gfx1250) — matrix operand order repaired and checked by reading, not compiled, not run on hardware
//
#include <hip/hip_runtime.h>

typedef __attribute__((ext_vector_type(16))) __bf16 v16bf;
typedef __attribute__((ext_vector_type(8)))  __bf16 v8bf;
typedef __attribute__((ext_vector_type(8)))  float  v8f;

#define Bdim 256
#define Tdim 512
#define Ddim 512
#define Hdim 512
#define NEMB 96
#define KCAT 1120   // 512 + 96 + 512
#define G4H  2048   // 4*H
#define ROWS 64     // bt rows staged per block in k_attn_scores
#define MT   4      // ROWS/16 M-tiles

// ---------- helpers ----------
__device__ __forceinline__ unsigned short f2bf(float f) {
    unsigned u = __builtin_bit_cast(unsigned, f);
    unsigned r = u + 0x7FFFu + ((u >> 16) & 1u);   // round-to-nearest-even
    return (unsigned short)(r >> 16);
}
__device__ __forceinline__ float sigm(float x) { return 1.f / (1.f + expf(-x)); }

__device__ __forceinline__ v16bf load2x8(const __bf16* p0, const __bf16* p1) {
    union { v16bf v; v8bf h[2]; } u;
    u.h[0] = *(const v8bf*)p0;
    u.h[1] = *(const v8bf*)p1;
    return u.v;
}

// ---------- fp32 -> bf16 convert ----------
__global__ void k_f2bf(const float* __restrict__ src, unsigned short* __restrict__ dst, int n) {
    int i = blockIdx.x * blockDim.x + threadIdx.x;
    if (i < n) dst[i] = f2bf(src[i]);
}

// ---------- generic bf16 WMMA GEMM: C[M,N] = A[M,K] * B[N,K]^T + bias ----------
// A,B row-major with K contiguous (lda=ldb=K). One wave per 16x16 tile.
__global__ __launch_bounds__(128) void k_gemm_bias_wmma(
    const __bf16* __restrict__ A, const __bf16* __restrict__ B,
    const float* __restrict__ bias1, const float* __restrict__ bias2,
    float* __restrict__ C, int M, int N, int K)
{
    const int tile = blockIdx.x * 4 + (threadIdx.x >> 5);
    const int ntiles = N >> 4;
    const int mt = tile / ntiles, nt = tile % ntiles;
    const int lane = threadIdx.x & 31;
    const int l16 = lane & 15, half = lane >> 4;
    const int ka = half * 8;       // A chunk base within 32-K tile
    const int kb = half * 8;       // B chunk base within 32-K tile

    const __bf16* aRow = A + (size_t)(mt * 16 + l16) * K;
    const __bf16* bRow = B + (size_t)(nt * 16 + l16) * K;

    v8f acc = {0.f, 0.f, 0.f, 0.f, 0.f, 0.f, 0.f, 0.f};
    const int nk = K >> 5;
    for (int kt = 0; kt < nk; ++kt) {
        const int k0 = kt * 32;
        v16bf a  = load2x8(aRow + k0 + ka, aRow + k0 + ka + 16);
        v16bf bm = load2x8(bRow + k0 + kb, bRow + k0 + kb + 16);
        acc = __builtin_amdgcn_wmma_f32_16x16x32_bf16(false, a, false, bm,
                                                      (short)0, acc, false, false);
    }
    const int n = nt * 16 + l16;
    float bv = 0.f;
    if (bias1) bv += bias1[n];
    if (bias2) bv += bias2[n];
#pragma unroll
    for (int v = 0; v < 8; ++v) {
        int m = mt * 16 + v + half * 8;
        C[(size_t)m * N + n] = acc[v] + bv;
    }
}

// ---------- fused attention scores ----------
// e[bt] = sum_h Wscore[h] * tanh( (batch_H[bt,:] . Wi2h[h,:]) + php[b,h] )
// One block = 64 bt rows staged in LDS as bf16; 4 waves x 8 N-tiles; each
// B fetch is reused across the 4 M-tiles (4 back-to-back WMMAs per B load).
__global__ __launch_bounds__(128) void k_attn_scores(
    const float* __restrict__ batchH, const __bf16* __restrict__ wi2h,
    const float* __restrict__ php, const float* __restrict__ wscore,
    float* __restrict__ e)
{
    __shared__ __align__(16) unsigned short aTile[ROWS * 520];  // bf16, padded stride
    __shared__ float eLds[ROWS];

    const int tid = threadIdx.x;
    const int row0 = blockIdx.x * ROWS;          // bt row base (T%64==0 -> same b)
    const int b = row0 / Tdim;

    if (tid < ROWS) eLds[tid] = 0.f;

    // stage 64x512 fp32 -> bf16 into LDS (float2 -> packed pair)
    const float* src = batchH + (size_t)row0 * Ddim;
    unsigned* lds32 = (unsigned*)aTile;
    for (int p = tid; p < ROWS * 256; p += 128) {
        int r = p >> 8, c2 = p & 255;
        float2 v = *(const float2*)(src + (size_t)r * Ddim + c2 * 2);
        lds32[r * 260 + c2] = (unsigned)f2bf(v.x) | ((unsigned)f2bf(v.y) << 16);
    }
    __syncthreads();

    const int wave = tid >> 5, lane = tid & 31;
    const int l16 = lane & 15, half = lane >> 4;
    const int ka = half * 8, kb = half * 8;
    const __bf16* aBase = (const __bf16*)aTile + l16 * 520;   // M-tile m adds m*16*520

    float er[MT][8];
#pragma unroll
    for (int m = 0; m < MT; ++m)
#pragma unroll
        for (int v = 0; v < 8; ++v) er[m][v] = 0.f;

    for (int nt = 0; nt < 8; ++nt) {
        const int n0 = (wave * 8 + nt) * 16;
        const __bf16* bRow = wi2h + (size_t)(n0 + l16) * Ddim;
        v8f acc[MT];
#pragma unroll
        for (int m = 0; m < MT; ++m)
            acc[m] = (v8f){0.f, 0.f, 0.f, 0.f, 0.f, 0.f, 0.f, 0.f};
#pragma unroll
        for (int kt = 0; kt < 16; ++kt) {
            const int k0 = kt * 32;
            v16bf bm = load2x8(bRow + k0 + kb, bRow + k0 + kb + 16);
#pragma unroll
            for (int m = 0; m < MT; ++m) {
                const __bf16* aRow = aBase + m * 16 * 520;
                v16bf a = load2x8(aRow + k0 + ka, aRow + k0 + ka + 16);
                acc[m] = __builtin_amdgcn_wmma_f32_16x16x32_bf16(false, a, false, bm,
                                                                 (short)0, acc[m], false, false);
            }
        }
        const int hN = n0 + l16;                 // output column index (C layout: N = lane%16)
        const float phpv = php[b * Hdim + hN];
        const float wsv  = wscore[hN];
#pragma unroll
        for (int m = 0; m < MT; ++m)
#pragma unroll
            for (int v = 0; v < 8; ++v)          // C layout: VGPR v -> M = v + 8*half
                er[m][v] += wsv * tanhf(acc[m][v] + phpv);
    }

    // reduce across the 16 lanes of each half-wave (same M rows), then across waves
#pragma unroll
    for (int m = 0; m < MT; ++m) {
#pragma unroll
        for (int v = 0; v < 8; ++v) {
            float s = er[m][v];
            s += __shfl_xor(s, 1); s += __shfl_xor(s, 2);
            s += __shfl_xor(s, 4); s += __shfl_xor(s, 8);
            er[m][v] = s;
        }
    }
    if (l16 == 0) {
#pragma unroll
        for (int m = 0; m < MT; ++m)
#pragma unroll
            for (int v = 0; v < 8; ++v)
                atomicAdd(&eLds[m * 16 + v + half * 8], er[m][v]);
    }
    __syncthreads();
    if (tid < ROWS) e[row0 + tid] = eLds[tid];
}

// ---------- softmax over T per batch ----------
__global__ __launch_bounds__(512) void k_softmax(const float* __restrict__ e,
                                                 float* __restrict__ alpha) {
    __shared__ float buf[512];
    const int b = blockIdx.x, t = threadIdx.x;
    const float v = e[b * Tdim + t];
    buf[t] = v; __syncthreads();
    for (int s = 256; s > 0; s >>= 1) { if (t < s) buf[t] = fmaxf(buf[t], buf[t + s]); __syncthreads(); }
    const float m = buf[0]; __syncthreads();
    const float ex = expf(v - m);
    buf[t] = ex; __syncthreads();
    for (int s = 256; s > 0; s >>= 1) { if (t < s) buf[t] += buf[t + s]; __syncthreads(); }
    alpha[b * Tdim + t] = ex / buf[0];
}

// ---------- context = alpha^T . batch_H ----------
__global__ __launch_bounds__(512) void k_context(const float* __restrict__ alpha,
                                                 const float* __restrict__ batchH,
                                                 float* __restrict__ ctx) {
    __shared__ float al[512];
    const int b = blockIdx.x, d = threadIdx.x;
    al[d] = alpha[b * Tdim + d];
    __syncthreads();
    const float* base = batchH + (size_t)b * Tdim * Ddim + d;
    float acc = 0.f;
#pragma unroll 8
    for (int t = 0; t < Tdim; ++t) acc += al[t] * base[(size_t)t * Ddim];
    ctx[b * Ddim + d] = acc;
}

// ---------- build concatenated bf16 operands for gate GEMM ----------
__global__ void k_build_xcat(const float* __restrict__ ctx, const float* __restrict__ oh,
                             const float* __restrict__ h, unsigned short* __restrict__ xcat) {
    int i = blockIdx.x * blockDim.x + threadIdx.x;
    if (i >= Bdim * KCAT) return;
    int b = i / KCAT, c = i % KCAT;
    float v = (c < Ddim) ? ctx[b * Ddim + c]
            : (c < Ddim + NEMB) ? oh[b * NEMB + (c - Ddim)]
            : h[b * Hdim + (c - Ddim - NEMB)];
    xcat[i] = f2bf(v);
}

__global__ void k_build_wcat(const float* __restrict__ Wih, const float* __restrict__ Whh,
                             unsigned short* __restrict__ wcat) {
    int i = blockIdx.x * blockDim.x + threadIdx.x;
    if (i >= G4H * KCAT) return;
    int j = i / KCAT, c = i % KCAT;
    float v = (c < Ddim + NEMB) ? Wih[j * (Ddim + NEMB) + c]
                                : Whh[j * Hdim + (c - Ddim - NEMB)];
    wcat[i] = f2bf(v);
}

// ---------- LSTM finalize: gate order i,f,g,o ----------
__global__ void k_lstm_final(const float* __restrict__ gates, const float* __restrict__ c,
                             float* __restrict__ out) {
    int i = blockIdx.x * blockDim.x + threadIdx.x;
    if (i >= Bdim * Hdim) return;
    int b = i >> 9, j = i & 511;
    const float* g = gates + (size_t)b * G4H;
    float ig = sigm(g[j]);
    float fg = sigm(g[Hdim + j]);
    float gg = tanhf(g[2 * Hdim + j]);
    float og = sigm(g[3 * Hdim + j]);
    float cn = fg * c[i] + ig * gg;
    float hn = og * tanhf(cn);
    out[i] = hn;                     // h_new
    out[Bdim * Hdim + i] = cn;       // c_new
}

extern "C" void kernel_launch(void* const* d_in, const int* in_sizes, int n_in,
                              void* d_out, int out_size, void* d_ws, size_t ws_size,
                              hipStream_t stream) {
    const float* h      = (const float*)d_in[0];
    const float* c      = (const float*)d_in[1];
    const float* bH     = (const float*)d_in[2];
    const float* oh     = (const float*)d_in[3];
    const float* Wi2h   = (const float*)d_in[4];
    const float* Wh2h   = (const float*)d_in[5];
    const float* bh2h   = (const float*)d_in[6];
    const float* Wscore = (const float*)d_in[7];
    const float* W_ih   = (const float*)d_in[8];
    const float* W_hh   = (const float*)d_in[9];
    const float* b_ih   = (const float*)d_in[10];
    const float* b_hh   = (const float*)d_in[11];
    float* out = (float*)d_out;

    char* w = (char*)d_ws;
    unsigned short* wi2h_bf = (unsigned short*)(w);            // 512*512 bf16
    unsigned short* h_bf    = (unsigned short*)(w + 524288);   // 256*512 bf16
    unsigned short* wh2h_bf = (unsigned short*)(w + 786432);   // 512*512 bf16
    float*          php     = (float*)(w + 1310720);           // 256*512 f32
    float*          e       = (float*)(w + 1835008);           // 256*512 f32
    float*          ctx     = (float*)(w + 2359296);           // 256*512 f32
    unsigned short* xcat    = (unsigned short*)(w + 2883584);  // 256*1120 bf16
    unsigned short* wcat    = (unsigned short*)(w + 3457024);  // 2048*1120 bf16
    float*          gates   = (float*)(w + 8044544);           // 256*2048 f32

    float* alpha = out + 2 * Bdim * Hdim;   // alpha written straight to output

    // conversions
    k_f2bf<<<(Hdim * Ddim + 255) / 256, 256, 0, stream>>>(Wi2h, wi2h_bf, Hdim * Ddim);
    k_f2bf<<<(Bdim * Hdim + 255) / 256, 256, 0, stream>>>(h, h_bf, Bdim * Hdim);
    k_f2bf<<<(Hdim * Hdim + 255) / 256, 256, 0, stream>>>(Wh2h, wh2h_bf, Hdim * Hdim);

    // php = h @ Wh2h^T + bh2h   (M=256, N=512, K=512) -> 512 tiles / 4 waves
    k_gemm_bias_wmma<<<128, 128, 0, stream>>>((const __bf16*)h_bf, (const __bf16*)wh2h_bf,
                                              bh2h, nullptr, php, Bdim, Hdim, Hdim);

    // fused proj + tanh + score   (BT/64 = 2048 blocks)
    k_attn_scores<<<(Bdim * Tdim) / ROWS, 128, 0, stream>>>(bH, (const __bf16*)wi2h_bf,
                                                            php, Wscore, e);

    k_softmax<<<Bdim, 512, 0, stream>>>(e, alpha);
    k_context<<<Bdim, 512, 0, stream>>>(alpha, bH, ctx);

    k_build_xcat<<<(Bdim * KCAT + 255) / 256, 256, 0, stream>>>(ctx, oh, h, xcat);
    k_build_wcat<<<(G4H * KCAT + 255) / 256, 256, 0, stream>>>(W_ih, W_hh, wcat);

    // gates = xcat @ wcat^T + b_ih + b_hh  (M=256, N=2048, K=1120) -> 2048 tiles / 4 waves
    k_gemm_bias_wmma<<<512, 128, 0, stream>>>((const __bf16*)xcat, (const __bf16*)wcat,
                                              b_ih, b_hh, gates, Bdim, G4H, KCAT);

    k_lstm_final<<<(Bdim * Hdim + 255) / 256, 256, 0, stream>>>(gates, c, out);
}
